// GroupedQueryAttention_33792802685286
// MI455X (gfx1250) — matrix operand order repaired and checked by reading, not compiled, not run on hardware
//
#include <hip/hip_runtime.h>
#include <hip/hip_bf16.h>

// ---------------------------------------------------------------------------
// Types for CDNA5 WMMA (wave32, v_wmma_f32_16x16x32_bf16)
// ---------------------------------------------------------------------------
typedef __attribute__((ext_vector_type(16))) __bf16 v16bf;
typedef __attribute__((ext_vector_type(8)))  __bf16 v8bf;
typedef __attribute__((ext_vector_type(8)))  float  v8f;
typedef __attribute__((ext_vector_type(4)))  unsigned int v4u;
typedef __attribute__((ext_vector_type(8)))  int v8i;
typedef __attribute__((ext_vector_type(4)))  int v4i;

union AFrag {              // assemble 16x32 bf16 A-fragment from two 8-half chunks
    v16bf v;
    v8bf  h[2];
};

union Pack8 {              // 8 bf16 <-> uint4 for vector copies with rescale
    uint4  u;
    __bf16 h[8];
};

__device__ __forceinline__ v8f wmma_bf16(v16bf a, v16bf b, v8f c) {
    // (neg_a, A, neg_b, B, c_mod, C, reuse_a, reuse_b)
    return __builtin_amdgcn_wmma_f32_16x16x32_bf16(false, a, false, b, (short)0, c,
                                                   false, false);
}

__device__ __forceinline__ v8f vzero8() {
    v8f z = {0.f, 0.f, 0.f, 0.f, 0.f, 0.f, 0.f, 0.f};
    return z;
}

__device__ __forceinline__ unsigned pack_bf16x2(float a, float b) {
    union { __bf16 h[2]; unsigned u; } x;
    x.h[0] = (__bf16)a;
    x.h[1] = (__bf16)b;
    return x.u;            // lowers to v_cvt_pk_bf16_f32
}

// ---------------------------------------------------------------------------
// TDM: issue a 2-D tile load (bf16 elements) via TENSOR_LOAD_TO_LDS.
// Descriptor layout per CDNA5 ISA.
//   tensor: [tensor_d1 rows x tensor_d0 cols], row stride = stride0 (elements)
//   tile:   [tile_d1 rows x tile_d0 cols], written contiguously to LDS
// 6-arg builtin form (clang-23 / therock-10.0 toolchain).
// ---------------------------------------------------------------------------
__device__ __forceinline__ void tdm_load_2d(unsigned lds_addr, const void* gaddr,
                                            unsigned tensor_d0, unsigned tensor_d1,
                                            unsigned tile_d0, unsigned tile_d1,
                                            unsigned stride0) {
    const unsigned long long ga = (unsigned long long)(uintptr_t)gaddr;
    v4u g0;
    g0.x = 1u;                                             // count=1, user descriptor
    g0.y = lds_addr;                                       // [63:32] LDS byte address
    g0.z = (unsigned)(ga & 0xFFFFFFFFu);                   // global addr [95:64]
    g0.w = (unsigned)((ga >> 32) & 0x01FFFFFFu)            // global addr hi
         | 0x80000000u;                                    // type=2 ("image") [127:126]
    v8i g1;
    g1[0] = (int)(1u << 16);                               // wg_mask=0, data_size=1 (2B)
    g1[1] = (int)((tensor_d0 & 0xFFFFu) << 16);            // tensor_dim0 [79:48] lo
    g1[2] = (int)((tensor_d0 >> 16) | ((tensor_d1 & 0xFFFFu) << 16)); // dim0 hi, dim1 lo
    g1[3] = (int)((tensor_d1 >> 16) | (tile_d0 << 16));    // dim1 hi, tile_dim0 [127:112]
    g1[4] = (int)(tile_d1 & 0xFFFFu);                      // tile_dim1, tile_dim2=0
    g1[5] = (int)stride0;                                  // tensor_dim0_stride lo
    g1[6] = 0;                                             // stride0 hi, stride1 lo
    g1[7] = 0;                                             // stride1 hi
    v4i g2 = {0, 0, 0, 0};
    v4i g3 = {0, 0, 0, 0};
    v8i g4 = {0, 0, 0, 0, 0, 0, 0, 0};
    __builtin_amdgcn_tensor_load_to_lds(g0, g1, g2, g3, g4, 0);
}

// ---------------------------------------------------------------------------
// Kernel 1: projection GEMM   C[M,N](bf16) = A[M,K](f32) * W[K,N](f32)
// Tile 128x128, K-step 64. 256 threads = 8 waves (4 x 2), wave tile 32x64.
// fp32 -> bf16 conversion happens while filling LDS (packed b32 stores for A).
// ---------------------------------------------------------------------------
__global__ void proj_gemm_bf16(const float* __restrict__ A,
                               const float* __restrict__ W,
                               __bf16* __restrict__ C,
                               int M, int N, int K) {
    __shared__ __bf16 As[128][64] __attribute__((aligned(64)));  // 16 KB
    __shared__ __bf16 Bt[128][64] __attribute__((aligned(64)));  // 16 KB, W tile transposed [n][k]

    const int tid  = threadIdx.x;
    const int wave = tid >> 5;
    const int lane = tid & 31;
    const int wm   = wave >> 1;           // 0..3 : 32-row strips
    const int wn   = wave & 1;            // 0..1 : 64-col strips
    const int mBase = blockIdx.x * 128;
    const int nBase = blockIdx.y * 128;

    const int laneN  = lane & 15;
    const int hiHalf = lane >> 4;

    v8f acc[2][4];
#pragma unroll
    for (int mt = 0; mt < 2; ++mt)
#pragma unroll
        for (int nt = 0; nt < 4; ++nt) acc[mt][nt] = vzero8();

    for (int k0 = 0; k0 < K; k0 += 64) {
        __syncthreads();
        {   // A tile: 128x64 f32 -> bf16 ; 32 elems / thread, packed pair stores
            const int row = tid >> 1;
            const int c0  = (tid & 1) * 32;
            const float* src = A + (size_t)(mBase + row) * K + k0 + c0;
            unsigned* dst = (unsigned*)&As[row][c0];
#pragma unroll
            for (int i = 0; i < 16; ++i)
                dst[i] = pack_bf16x2(src[2 * i], src[2 * i + 1]);
        }
        {   // W tile: 64x128 f32 -> bf16, stored transposed (2-byte scatter)
            const int kk = tid >> 2;
            const int c0 = (tid & 3) * 32;
            const float* src = W + (size_t)(k0 + kk) * N + nBase + c0;
#pragma unroll
            for (int i = 0; i < 32; ++i) Bt[c0 + i][kk] = (__bf16)src[i];
        }
        __syncthreads();

#pragma unroll
        for (int kc = 0; kc < 64; kc += 32) {
            // A fragments: row = lane&15, 16 contiguous K at hi*16
            AFrag a[2];
            const int ak = kc + hiHalf * 16;
#pragma unroll
            for (int mt = 0; mt < 2; ++mt) {
                const int arow = wm * 32 + mt * 16 + laneN;
                a[mt].h[0] = *(const v8bf*)&As[arow][ak];
                a[mt].h[1] = *(const v8bf*)&As[arow][ak + 8];
            }
            // B fragments: col = lane&15, 16 contiguous K at hi*16
            const int bk = kc + hiHalf * 16;
#pragma unroll
            for (int nt = 0; nt < 4; ++nt) {
                const int bcol = wn * 64 + nt * 16 + laneN;
                v16bf b = *(const v16bf*)&Bt[bcol][bk];
#pragma unroll
                for (int mt = 0; mt < 2; ++mt)
                    acc[mt][nt] = wmma_bf16(a[mt].v, b, acc[mt][nt]);
            }
        }
    }

    // C layout: VGPR r -> row r (lanes 0-15) / row r+8 (lanes 16-31), col = lane&15
#pragma unroll
    for (int mt = 0; mt < 2; ++mt)
#pragma unroll
        for (int nt = 0; nt < 4; ++nt)
#pragma unroll
            for (int r = 0; r < 8; ++r) {
                const int row = mBase + wm * 32 + mt * 16 + r + hiHalf * 8;
                const int col = nBase + wn * 64 + nt * 16 + laneN;
                C[(size_t)row * N + col] = (__bf16)acc[mt][nt][r];
            }
}

// ---------------------------------------------------------------------------
// Kernel 2: V transpose per head:  Vt[bh][d][s] = Vp[bh][s][d]   (bf16)
// ---------------------------------------------------------------------------
__global__ void transpose_v(const __bf16* __restrict__ Vp,
                            __bf16* __restrict__ Vt, int S) {
    __shared__ __bf16 tile[64][65];
    const int bh    = blockIdx.z;
    const int sBase = blockIdx.x * 64;
    const int tid   = threadIdx.x;
#pragma unroll
    for (int i = 0; i < 16; ++i) {
        const int idx = tid + i * 256;
        const int s = idx >> 6, d = idx & 63;
        tile[d][s] = Vp[((size_t)bh * S + sBase + s) * 64 + d];
    }
    __syncthreads();
#pragma unroll
    for (int i = 0; i < 16; ++i) {
        const int idx = tid + i * 256;
        const int d = idx >> 6, s = idx & 63;
        Vt[((size_t)bh * 64 + d) * S + sBase + s] = tile[d][s];
    }
}

// ---------------------------------------------------------------------------
// Kernel 3: flash attention, bf16 WMMA, online softmax.
// Grid: (S/128, B*H). Block: 256 threads = 8 waves; wave w owns 16 query rows.
// K / V^T tiles are streamed into LDS by the Tensor Data Mover (wave 0 issues
// the descriptors, waits TENSORcnt==0, then the workgroup barrier publishes).
// ---------------------------------------------------------------------------
#define HD 64
#define QB 128
#define KBLK 64

__global__ void flash_attn(const __bf16* __restrict__ Qp,
                           const __bf16* __restrict__ Kp,
                           const __bf16* __restrict__ Vt,
                           float* __restrict__ out,
                           int Bn, int H, int S) {
    __shared__ __bf16 Qs[QB][HD]        __attribute__((aligned(64))); // 16 KB
    __shared__ __bf16 Ks[KBLK][HD]      __attribute__((aligned(64))); //  8 KB
    __shared__ __bf16 Vs[HD][KBLK]      __attribute__((aligned(64))); //  8 KB (V^T tile)
    __shared__ __bf16 Ps[8][16][KBLK]   __attribute__((aligned(64))); // 16 KB

    const int tid    = threadIdx.x;
    const int wave   = tid >> 5;
    const int lane   = tid & 31;
    const int laneN  = lane & 15;
    const int hiHalf = lane >> 4;
    const int qBase  = blockIdx.x * QB;
    const int bh     = blockIdx.y;
    const size_t rowBase = (size_t)bh * S;   // head-row base into flat [BH*S][64]

    {   // load Q tile (128x64 bf16), pre-scaled by 1/sqrt(DIM) = 1/32 (exact in bf16)
        const int row = tid >> 1;
        const int c0  = (tid & 1) * 32;
        const uint4* src = (const uint4*)(Qp + (rowBase + qBase + row) * HD + c0);
        uint4* dst = (uint4*)&Qs[row][c0];
#pragma unroll
        for (int j = 0; j < 4; ++j) {
            Pack8 p;
            p.u = src[j];
#pragma unroll
            for (int i = 0; i < 8; ++i)
                p.h[i] = (__bf16)((float)p.h[i] * 0.03125f);
            dst[j] = p.u;
        }
    }

    const unsigned ldsK = (unsigned)(uintptr_t)(void*)&Ks[0][0];
    const unsigned ldsV = (unsigned)(uintptr_t)(void*)&Vs[0][0];

    v8f O[4];
#pragma unroll
    for (int t = 0; t < 4; ++t) O[t] = vzero8();
    float m_r[8], l_r[8];
#pragma unroll
    for (int r = 0; r < 8; ++r) { m_r[r] = -3.0e38f; l_r[r] = 0.f; }

    for (int kb = 0; kb < S; kb += KBLK) {
        __syncthreads();                 // previous K/V tiles fully consumed
        if (wave == 0) {
            // K tile: rows = keys (64), cols = hd (64), row stride = HD
            tdm_load_2d(ldsK, Kp + (rowBase + kb) * HD,
                        HD, (unsigned)S, HD, KBLK, HD);
            // V^T tile: rows = d' (64), cols = keys (64), row stride = S
            tdm_load_2d(ldsV, Vt + ((size_t)bh * HD) * S + kb,
                        (unsigned)S, HD, KBLK, HD, (unsigned)S);
            __builtin_amdgcn_s_wait_tensorcnt(0);
        }
        __syncthreads();                 // tiles visible to all 8 waves

        // ---- scores: Sc[16q x 64k] = Q(16x64) * K^T, 8 WMMAs
        AFrag a0, a1;
        const int qrow = wave * 16 + laneN;
        const int ak = hiHalf * 16;
        a0.h[0] = *(const v8bf*)&Qs[qrow][ak];
        a0.h[1] = *(const v8bf*)&Qs[qrow][ak + 8];
        a1.h[0] = *(const v8bf*)&Qs[qrow][32 + ak];
        a1.h[1] = *(const v8bf*)&Qs[qrow][32 + ak + 8];

        v8f Sc[4] = {vzero8(), vzero8(), vzero8(), vzero8()};
        const int bk = hiHalf * 16;
#pragma unroll
        for (int nt = 0; nt < 4; ++nt) {
            const int key = nt * 16 + laneN;
            v16bf b0 = *(const v16bf*)&Ks[key][bk];        // hd 0..31
            v16bf b1 = *(const v16bf*)&Ks[key][32 + bk];   // hd 32..63
            Sc[nt] = wmma_bf16(a0.v, b0, Sc[nt]);
            Sc[nt] = wmma_bf16(a1.v, b1, Sc[nt]);
        }

        // ---- online softmax (row r lives in lanes 0-15 for M=r, 16-31 for M=r+8)
#pragma unroll
        for (int r = 0; r < 8; ++r) {
            float rm = fmaxf(fmaxf(Sc[0][r], Sc[1][r]), fmaxf(Sc[2][r], Sc[3][r]));
            rm = fmaxf(rm, __shfl_xor(rm, 1, 32));
            rm = fmaxf(rm, __shfl_xor(rm, 2, 32));
            rm = fmaxf(rm, __shfl_xor(rm, 4, 32));
            rm = fmaxf(rm, __shfl_xor(rm, 8, 32));
            const float mn = fmaxf(m_r[r], rm);
            const float cf = __expf(m_r[r] - mn);
            float p[4];
            float rs = 0.f;
#pragma unroll
            for (int nt = 0; nt < 4; ++nt) {
                p[nt] = __expf(Sc[nt][r] - mn);
                rs += p[nt];
            }
            rs += __shfl_xor(rs, 1, 32);
            rs += __shfl_xor(rs, 2, 32);
            rs += __shfl_xor(rs, 4, 32);
            rs += __shfl_xor(rs, 8, 32);
            l_r[r] = l_r[r] * cf + rs;
            m_r[r] = mn;
#pragma unroll
            for (int t = 0; t < 4; ++t) O[t][r] *= cf;
            const int prow = r + hiHalf * 8;
#pragma unroll
            for (int nt = 0; nt < 4; ++nt)
                Ps[wave][prow][nt * 16 + laneN] = (__bf16)p[nt];
        }
        __syncthreads();

        // ---- O += P(16x64) * V(64x16) per 16-wide d' tile, 8 WMMAs
        AFrag ap0, ap1;
        const int pk = hiHalf * 16;
        ap0.h[0] = *(const v8bf*)&Ps[wave][laneN][pk];
        ap0.h[1] = *(const v8bf*)&Ps[wave][laneN][pk + 8];
        ap1.h[0] = *(const v8bf*)&Ps[wave][laneN][32 + pk];
        ap1.h[1] = *(const v8bf*)&Ps[wave][laneN][32 + pk + 8];
#pragma unroll
        for (int t = 0; t < 4; ++t) {
            const int dcol = t * 16 + laneN;
            v16bf bv0 = *(const v16bf*)&Vs[dcol][bk];        // keys 0..31
            v16bf bv1 = *(const v16bf*)&Vs[dcol][32 + bk];   // keys 32..63
            O[t] = wmma_bf16(ap0.v, bv0, O[t]);
            O[t] = wmma_bf16(ap1.v, bv1, O[t]);
        }
    }

    // ---- epilogue: out[b, sq, h*64 + d'] = O / l   (fp32, transposed head layout)
    const int b = bh / H;
    const int h = bh % H;
    const int D = H * HD;
#pragma unroll
    for (int t = 0; t < 4; ++t)
#pragma unroll
        for (int r = 0; r < 8; ++r) {
            const int sq = qBase + wave * 16 + r + hiHalf * 8;
            const int dc = t * 16 + laneN;
            out[((size_t)b * S + sq) * D + h * HD + dc] = O[t][r] / l_r[r];
        }
}

// ---------------------------------------------------------------------------
// Launch
// ---------------------------------------------------------------------------
extern "C" void kernel_launch(void* const* d_in, const int* in_sizes, int n_in,
                              void* d_out, int out_size, void* d_ws, size_t ws_size,
                              hipStream_t stream) {
    const float* q  = (const float*)d_in[0];
    const float* k  = (const float*)d_in[1];
    const float* v  = (const float*)d_in[2];
    const float* wq = (const float*)d_in[3];
    const float* wk = (const float*)d_in[4];
    const float* wv = (const float*)d_in[5];
    float* out = (float*)d_out;

    const int Bn = 2, S = 2048, D = 1024, H = 16;
    const int M = Bn * S;                 // 4096
    const size_t nProj = (size_t)M * D;   // 4M elements

    __bf16* projQ = (__bf16*)d_ws;
    __bf16* projK = projQ + nProj;
    __bf16* projV = projK + nProj;
    __bf16* vT    = projV + nProj;        // [BH][64][S]

    dim3 gGemm(M / 128, D / 128);
    proj_gemm_bf16<<<gGemm, 256, 0, stream>>>(q, wq, projQ, M, D, D);
    proj_gemm_bf16<<<gGemm, 256, 0, stream>>>(k, wk, projK, M, D, D);
    proj_gemm_bf16<<<gGemm, 256, 0, stream>>>(v, wv, projV, M, D, D);

    dim3 gT(S / 64, 1, Bn * H);
    transpose_v<<<gT, 256, 0, stream>>>(projV, vT, S);

    dim3 gA(S / QB, Bn * H);
    flash_attn<<<gA, 256, 0, stream>>>(projQ, projK, vT, out, Bn, H, S);
}
